// KGLayer_49082886259209
// MI455X (gfx1250) — matrix operand order repaired and checked by reading, not compiled, not run on hardware
//
#include <hip/hip_runtime.h>
#include <hip/hip_bf16.h>

// ---------------- CDNA5 WMMA types ----------------
typedef __attribute__((ext_vector_type(16))) __bf16          v16bf;
typedef __attribute__((ext_vector_type(8)))  float           v8f;
typedef __attribute__((ext_vector_type(8)))  unsigned short  u16x8;
typedef __attribute__((ext_vector_type(16))) unsigned short  u16x16;
typedef __attribute__((ext_vector_type(4)))  int             v4i;

#define WPB 8                 // waves per block (wave32) -> one 256-thread block per WGP
static constexpr int ROWB = 392;   // bf16 elems per staged row (384 + 8 pad -> no LDS bank conflicts)
static constexpr int CROW = 132;   // f32 elems per c-row (128 + 4 pad)
static constexpr int WELEMS = 8 * 12 * 32 * 16;   // 49152 packed bf16 weight elems (96 KB)

// --- probe-confirmed: CDNA5 async global->LDS copy (ASYNCcnt-tracked) ---
// signature: (addrspace(1) v4i* src, addrspace(3) v4i* lds, int offset, int cpol)
#if defined(__has_builtin)
#if __has_builtin(__builtin_amdgcn_global_load_async_to_lds_b128)
#define HAVE_ASYNC_LDS 1
#endif
#endif
#ifndef HAVE_ASYNC_LDS
#define HAVE_ASYNC_LDS 0
#endif

__device__ __forceinline__ unsigned short f2bf(float f) {
    unsigned u = __float_as_uint(f);
    u += 0x7FFFu + ((u >> 16) & 1u);      // round-to-nearest-even bf16
    return (unsigned short)(u >> 16);
}

// ---------------- zero scratch/output ----------------
__global__ void zero_kernel(float* __restrict__ p, long long n) {
    long long i = (long long)blockIdx.x * blockDim.x + threadIdx.x;
    long long stride = (long long)gridDim.x * blockDim.x;
    for (; i < n; i += stride) p[i] = 0.0f;
}

// ---------------- f32 -> bf16 table conversion ----------------
__global__ void cvt_bf16_kernel(const float* __restrict__ in, unsigned short* __restrict__ out,
                                long long n) {
    long long i = (long long)blockIdx.x * blockDim.x + threadIdx.x;
    long long stride = (long long)gridDim.x * blockDim.x;
    for (; i < n; i += stride) out[i] = f2bf(in[i]);
}

// ---------------- pack a_w into WMMA B-matrix lane layout (bf16) ----------------
// B[k][n] = a_w[n][k]. For 16x16x32 bf16 B (wave32): lane L holds col N=L%16,
// K = k0+8*(L/16)..+7 and k0+16+8*(L/16)..+7.
// Packed: wpk[(((nt*12)+kt)*32 + lane)*16 + j]
__global__ void pack_w_kernel(const float* __restrict__ aw, unsigned short* __restrict__ wpk) {
    int i = blockIdx.x * blockDim.x + threadIdx.x;
    if (i >= WELEMS) return;
    int j    = i & 15;
    int lane = (i >> 4) & 31;
    int blk  = i >> 9;            // (nt*12 + kt)
    int kt   = blk % 12;
    int nt   = blk / 12;
    int n = nt * 16 + (lane & 15);
    int k = kt * 32 + (j >> 3) * 16 + (lane >> 4) * 8 + (j & 7);
    wpk[i] = f2bf(aw[n * 384 + k]);
}

// ---------------- main edge kernel ----------------
// BF16TAB: gather from pre-converted bf16 tables (preferred; half the gather bytes)
template<bool BF16TAB>
__global__ __launch_bounds__(WPB * 32)
void edge_kernel(const float* __restrict__ ent, const float* __restrict__ rel_emb,
                 const unsigned short* __restrict__ entb, const unsigned short* __restrict__ relb,
                 const float* __restrict__ a_b, const float* __restrict__ a2_w,
                 const float* __restrict__ a2_b,
                 const int* __restrict__ src, const int* __restrict__ dst,
                 const int* __restrict__ rel,
                 const unsigned short* __restrict__ wpk,
                 float* __restrict__ out_ent, float* __restrict__ out_rel,
                 float* __restrict__ ebsum, float* __restrict__ relcnt,
                 int E, int nTiles)
{
    // CDNA5: 320 KB LDS per workgroup. 96 KB resident B-matrix + 8 waves x 2 x 12.25 KB
    // double-buffered staging = 299,008 B total.
    __shared__ unsigned short shw[WELEMS];               // 98304 B: packed bf16 weights
    __shared__ unsigned short sh[WPB][2][16 * ROWB];     // 200704 B staging / c-dump

    const int lane = threadIdx.x & 31;
    const int warp = threadIdx.x >> 5;
    const int hi   = lane >> 4;       // half-wave id
    const int lm   = lane & 15;

    // -------- one-time: stage packed weights into LDS (shared by all 8 waves) --------
    {
        const uint4* gw = (const uint4*)wpk;
        uint4* lw = (uint4*)shw;
        for (int i = threadIdx.x; i < WELEMS / 8; i += WPB * 32)   // 6144 uint4
            lw[i] = gw[i];
    }
    __syncthreads();   // executed exactly once by every wave -> safe before divergent loops

    const int wavesTotal = gridDim.x * WPB;
    const int wid = blockIdx.x * WPB + warp;
    const float a2b = a2_b[0];

    // issue the gather for tile t into staging buffer b (24 async b128 per lane)
    auto stageTile = [&](int t, int b) {
        unsigned short* sb = &sh[warp][b][0];
        const int bs = t * 16;
        if (BF16TAB) {
            for (int it = lane; it < 16 * 48; it += 32) {   // 48 x 16B chunks per row
                int r = it / 48;
                int c = it - r * 48;
                int k = c * 8;
                int e = bs + r; if (e >= E) e = E - 1;
                const unsigned short* p;
                if (k < 128)      p = entb + (size_t)src[e] * 128 + k;
                else if (k < 256) p = entb + (size_t)dst[e] * 128 + (k - 128);
                else              p = relb + (size_t)rel[e] * 128 + (k - 256);
#if HAVE_ASYNC_LDS
                __builtin_amdgcn_global_load_async_to_lds_b128(
                    (__attribute__((address_space(1))) v4i*)p,
                    (__attribute__((address_space(3))) v4i*)(sb + r * ROWB + k),
                    0, 0);
#else
                *(uint4*)(sb + r * ROWB + k) = *(const uint4*)p;
#endif
            }
        } else {
            // fallback: gather f32 and convert inline (96 float4 per row)
            for (int it = lane; it < 16 * 96; it += 32) {
                int r  = it / 96;
                int c4 = it - r * 96;
                int k  = c4 * 4;
                int e  = bs + r; if (e >= E) e = E - 1;
                const float* p;
                if (k < 128)      p = ent     + (size_t)src[e] * 128 + k;
                else if (k < 256) p = ent     + (size_t)dst[e] * 128 + (k - 128);
                else              p = rel_emb + (size_t)rel[e] * 128 + (k - 256);
                float4 v = *(const float4*)p;
                unsigned lo = (unsigned)f2bf(v.x) | ((unsigned)f2bf(v.y) << 16);
                unsigned hi2 = (unsigned)f2bf(v.z) | ((unsigned)f2bf(v.w) << 16);
                uint2 pk; pk.x = lo; pk.y = hi2;
                *(uint2*)(sb + r * ROWB + k) = pk;
            }
        }
    };

    int tile = wid;
    int p = 0;
    if (tile < nTiles) stageTile(tile, p);      // prologue prefetch

    for (; tile < nTiles; tile += wavesTotal, p ^= 1) {
        const int base = tile * 16;
        const int nxt = tile + wavesTotal;

        // cross-tile pipeline: issue next tile's gather into the other buffer
        // BEFORE waiting on the current batch (async loads complete in order).
        if (nxt < nTiles) stageTile(nxt, p ^ 1);

#if HAVE_ASYNC_LDS
        if (BF16TAB) {
            if (nxt < nTiles) asm volatile("s_wait_asynccnt 0x18" ::: "memory"); // <=24: current batch done
            else              asm volatile("s_wait_asynccnt 0x0"  ::: "memory");
        }
#endif
        asm volatile("s_wait_dscnt 0x0" ::: "memory");
        __builtin_amdgcn_wave_barrier();

        unsigned short* wb = &sh[warp][p][0];
        float* cd = (float*)wb;       // c-tile dump reuses the staging region (phase-separated)

        // -------- GEMM: c[16x128] = h[16x384] @ a_w^T, bf16 WMMA, f32 accum --------
        v8f acc[8];
        #pragma unroll
        for (int nt = 0; nt < 8; ++nt) acc[nt] = (v8f)(0.0f);

        for (int kt = 0; kt < 12; ++kt) {
            const int k0 = kt * 32;
            // A-frag (16-bit A layout): lane holds row M=lm; K = k0+8*hi..+7 and k0+16+8*hi..+7
            const unsigned short* ap = wb + lm * ROWB + k0 + 8 * hi;
            u16x8 alo = *(const u16x8*)ap;
            u16x8 ahi = *(const u16x8*)(ap + 16);
            u16x16 au = __builtin_shufflevector(alo, ahi,
                        0,1,2,3,4,5,6,7,8,9,10,11,12,13,14,15);
            v16bf afrag = __builtin_bit_cast(v16bf, au);
            // clause all 8 B-frag loads, then burst 8 WMMAs (overlap DS with XDL)
            u16x16 bu[8];
            #pragma unroll
            for (int nt = 0; nt < 8; ++nt)
                bu[nt] = *(const u16x16*)(shw + ((((nt * 12) + kt) * 32 + lane) << 4));
            #pragma unroll
            for (int nt = 0; nt < 8; ++nt) {
                acc[nt] = __builtin_amdgcn_wmma_f32_16x16x32_bf16(
                    false, afrag, false, __builtin_bit_cast(v16bf, bu[nt]),
                    (short)0, acc[nt], false, false);
            }
        }

        // -------- dump c (+bias) to LDS [16 x 128], padded rows --------
        asm volatile("s_wait_dscnt 0x0" ::: "memory");
        #pragma unroll
        for (int nt = 0; nt < 8; ++nt) {
            float bias = a_b[nt * 16 + lm];
            #pragma unroll
            for (int r = 0; r < 8; ++r) {
                // C layout: VGPR r -> row M = r + 8*hi, col N = nt*16 + lm
                cd[(r + 8 * hi) * CROW + nt * 16 + lm] = acc[nt][r] + bias;
            }
        }
        asm volatile("s_wait_dscnt 0x0" ::: "memory");
        __builtin_amdgcn_wave_barrier();

        // -------- attention: b = leaky_relu(c . a2_w + a2_b); e_b = exp(b) --------
        float part = 0.0f;
        const float* crow = cd + lm * CROW + hi * 64;
        const float* w2   = a2_w + hi * 64;
        #pragma unroll
        for (int o = 0; o < 64; o += 4) {
            float4 c4 = *(const float4*)(crow + o);
            part = fmaf(c4.x, w2[o + 0], part);
            part = fmaf(c4.y, w2[o + 1], part);
            part = fmaf(c4.z, w2[o + 2], part);
            part = fmaf(c4.w, w2[o + 3], part);
        }
        float full = part + __shfl_xor(part, 16, 32);
        float bv = full + a2b;
        bv = bv > 0.0f ? bv : 0.01f * bv;
        float eb = expf(bv);

        // -------- scatter: segment sums via L2 atomics --------
        int e = base + lm;
        if (e < E) {
            int sv = src[e];
            int rv = rel[e];
            if (hi == 0) atomicAdd(&ebsum[sv], eb);
            else         atomicAdd(&relcnt[rv], 1.0f);
            float* oe = out_ent + (size_t)sv * 128 + hi * 64;
            float* orl = out_rel + (size_t)rv * 128 + hi * 64;
            const float* cr = cd + lm * CROW + hi * 64;
            #pragma unroll 4
            for (int o = 0; o < 64; ++o) {
                float t = eb * cr[o];
                atomicAdd(oe + o, t);
                atomicAdd(orl + o, t);
            }
        }
    }
}

// ---------------- finalize: normalize + ELU ----------------
__global__ void finalize_kernel(float* __restrict__ out_ent, float* __restrict__ out_rel,
                                const float* __restrict__ ebsum, const float* __restrict__ relcnt,
                                int nEnt, int nRel)
{
    long long total = ((long long)nEnt + nRel) * 128;
    long long i = (long long)blockIdx.x * blockDim.x + threadIdx.x;
    long long stride = (long long)gridDim.x * blockDim.x;
    long long entTot = (long long)nEnt * 128;
    for (; i < total; i += stride) {
        if (i < entTot) {
            int n = (int)(i >> 7);
            float s = ebsum[n];
            float d = (s == 0.0f) ? 1e-12f : s;
            float x = out_ent[i] / d;
            out_ent[i] = x > 0.0f ? x : expm1f(x);
        } else {
            long long j = i - entTot;
            int r = (int)(j >> 7);
            float c = relcnt[r]; if (c < 1.0f) c = 1.0f;
            float x = out_rel[j] / c;
            out_rel[j] = x > 0.0f ? x : expm1f(x);
        }
    }
}

extern "C" void kernel_launch(void* const* d_in, const int* in_sizes, int n_in,
                              void* d_out, int out_size, void* d_ws, size_t ws_size,
                              hipStream_t stream) {
    const float* ent     = (const float*)d_in[0];
    const float* rel_emb = (const float*)d_in[1];
    const float* a_w     = (const float*)d_in[2];
    const float* a_b     = (const float*)d_in[3];
    const float* a2_w    = (const float*)d_in[4];
    const float* a2_b    = (const float*)d_in[5];
    const int*   src     = (const int*)d_in[6];
    const int*   dst     = (const int*)d_in[7];
    const int*   rel     = (const int*)d_in[8];

    const int E     = in_sizes[6];
    const int N_ENT = in_sizes[0] / 128;
    const int N_REL = in_sizes[1] / 128;

    float* out_ent = (float*)d_out;
    float* out_rel = out_ent + (size_t)N_ENT * 128;

    // workspace layout: wpk | ebsum | relcnt | (optional) bf16 ent table | bf16 rel table
    size_t off = (size_t)WELEMS * sizeof(unsigned short);
    unsigned short* wpk = (unsigned short*)d_ws;
    float* ebsum  = (float*)((char*)d_ws + off);   off += (size_t)N_ENT * 4;
    float* relcnt = (float*)((char*)d_ws + off);   off += (size_t)N_REL * 4;
    off = (off + 15) & ~(size_t)15;
    unsigned short* entb = (unsigned short*)((char*)d_ws + off);
    size_t entBytes = (size_t)N_ENT * 128 * 2;
    size_t relBytes = (size_t)N_REL * 128 * 2;
    unsigned short* relb = (unsigned short*)((char*)d_ws + off + entBytes);
    const bool useBf16Tab = (ws_size >= off + entBytes + relBytes);

    // 1) zero accumulators (d_out is poisoned by the harness each run)
    zero_kernel<<<2048, 256, 0, stream>>>((float*)d_out, (long long)out_size);
    zero_kernel<<<256, 256, 0, stream>>>(ebsum, (long long)(N_ENT + N_REL));

    // 2) pack weights; pre-convert embedding tables to bf16 (halves gather traffic)
    pack_w_kernel<<<(WELEMS + 255) / 256, 256, 0, stream>>>(a_w, wpk);
    if (useBf16Tab) {
        cvt_bf16_kernel<<<2048, 256, 0, stream>>>(ent, entb, (long long)N_ENT * 128);
        cvt_bf16_kernel<<<64, 256, 0, stream>>>(rel_emb, relb, (long long)N_REL * 128);
    }

    // 3) edge GEMM + attention + scatter
    int nTiles = (E + 15) / 16;
    int blocks = (nTiles + WPB - 1) / WPB;
    if (blocks > 1024) blocks = 1024;
    if (useBf16Tab) {
        edge_kernel<true><<<blocks, WPB * 32, 0, stream>>>(ent, rel_emb, entb, relb,
                                                           a_b, a2_w, a2_b, src, dst, rel, wpk,
                                                           out_ent, out_rel, ebsum, relcnt,
                                                           E, nTiles);
    } else {
        edge_kernel<false><<<blocks, WPB * 32, 0, stream>>>(ent, rel_emb, entb, relb,
                                                            a_b, a2_w, a2_b, src, dst, rel, wpk,
                                                            out_ent, out_rel, ebsum, relcnt,
                                                            E, nTiles);
    }

    // 4) normalize + ELU
    finalize_kernel<<<2048, 256, 0, stream>>>(out_ent, out_rel, ebsum, relcnt, N_ENT, N_REL);
}
